// SpaceAttention_16097537426195
// MI455X (gfx1250) — matrix operand order repaired and checked by reading, not compiled, not run on hardware
//
#include <hip/hip_runtime.h>
#include <hip/hip_bf16.h>

// Problem constants (from reference setup_inputs): B=16, F=8, P=196, D=768, H=12
#define BATCH   16
#define NF      8
#define NP      196
#define G_TOK   1569          // 1 + NF*NP
#define D_MODEL 768
#define NH      12
#define HD      64
#define BHEADS  (BATCH*NH)    // 192
#define MROWS   (BATCH*G_TOK) // 25104
#define QSCALE  0.125f        // 64^-0.5

typedef __attribute__((ext_vector_type(16))) __bf16     v16bf;
typedef __attribute__((ext_vector_type(8)))  float       v8f;
typedef __attribute__((ext_vector_type(4)))  unsigned    v4u;
typedef __attribute__((ext_vector_type(8)))  int         v8i;
typedef __attribute__((ext_vector_type(4)))  int         v4i;

__device__ __forceinline__ v8f wmma_bf16(v16bf a, v16bf b, v8f c) {
  return __builtin_amdgcn_wmma_f32_16x16x32_bf16(
      false, a, false, b, (short)0, c, false, false);
}

// ---------------------------------------------------------------------------
// TDM: async 2-D bf16 tile load, global -> LDS, with 4B pad after every 64B
// row (LDS row pitch = 68B = 17 dwords, coprime with 64 banks).
// Descriptor per CDNA5 ISA ch.8. tile_cols fixed at 32 elems (64 bytes).
// ---------------------------------------------------------------------------
#define TILE_KE   32            // staged K elements per step (64 bytes/row)
#define LDSROW_E  34            // bf16 elems per padded LDS row (68 bytes)

__device__ __forceinline__ void tdm_load_2d(unsigned lds_off, const __bf16* gptr,
                                            unsigned rows_remaining,
                                            unsigned tile_rows,
                                            unsigned cols_remaining,
                                            unsigned row_stride_elems) {
  const unsigned long long ga = (unsigned long long)(uintptr_t)gptr;
  v4u g0;
  g0[0] = 1u;                                        // count=1, user mode
  g0[1] = lds_off;                                   // LDS byte offset
  g0[2] = (unsigned)(ga & 0xFFFFFFFFu);              // global_addr[31:0]
  g0[3] = (unsigned)((ga >> 32) & 0x1FFFFFFu)        // global_addr[56:32]
          | (2u << 30);                              // type=2 ("image")
  const unsigned td0 = cols_remaining;               // tensor_dim0 (elems)
  const unsigned td1 = rows_remaining;               // tensor_dim1 (rows)
  const unsigned long long st0 = row_stride_elems;   // tensor_dim0_stride
  v8i g1;
  g1[0] = (int)((1u << 16)                           // data_size = 2 bytes
              | (1u << 20)                           // pad_enable
              | (3u << 22));                         // pad_interval: 64B; pad_amount=1dw
  g1[1] = (int)((td0 & 0xFFFFu) << 16);              // tensor_dim0[15:0] @bit48
  g1[2] = (int)((td0 >> 16) | ((td1 & 0xFFFFu) << 16));
  g1[3] = (int)((td1 >> 16) | ((unsigned)TILE_KE << 16));  // tile_dim0
  g1[4] = (int)tile_rows;                            // tile_dim1 ; tile_dim2=0
  g1[5] = (int)(st0 & 0xFFFFFFFFu);
  g1[6] = (int)((st0 >> 32) & 0xFFFFu);              // dim1_stride = 0
  g1[7] = 0;
  const v4i z4 = {0, 0, 0, 0};
#if defined(__clang_major__) && (__clang_major__ >= 23)
  const v8i z8 = {0, 0, 0, 0, 0, 0, 0, 0};
  __builtin_amdgcn_tensor_load_to_lds(g0, g1, z4, z4, z8, 0);
#else
  __builtin_amdgcn_tensor_load_to_lds(g0, g1, z4, z4, 0);
#endif
}

// ---------------------------------------------------------------------------
// Kernel 0: f32 -> bf16 conversion (n divisible by 4).
// ---------------------------------------------------------------------------
__global__ __launch_bounds__(256)
void cvt_bf16_kernel(const float* __restrict__ src, __bf16* __restrict__ dst, int n4) {
  const int i = blockIdx.x * 256 + threadIdx.x;
  if (i < n4) {
    const float4 v = ((const float4*)src)[i];
    __bf16* o = dst + (size_t)i * 4;
    o[0] = (__bf16)v.x; o[1] = (__bf16)v.y; o[2] = (__bf16)v.z; o[3] = (__bf16)v.w;
  }
}

// ---------------------------------------------------------------------------
// Double-buffered TDM-staged bf16 GEMM body (shared by kernels 1 & 4).
// Block: 256 thr = 8 waves (wm = wave>>2 in 0..1, wn = wave&3).
// Tile: M=32, N=256, K staged in steps of 32 elems; ping/pong unrolled so
// LDS buffer offsets are immediates; s_wait_tensorcnt executed by ALL waves
// (no-op for non-issuers; issuer gates the following barrier).
// LDS: A bufs @ {0, 2240}, B bufs @ {4480, 21952}; total 39424 B.
// ---------------------------------------------------------------------------
#define A_OFF0 0u
#define A_OFF1 2240u
#define B_OFF0 4480u
#define B_OFF1 21952u
#define GEMM_LDS_BYTES 39424

__device__ __forceinline__ void gemm_step(const char* smem, unsigned aoff, unsigned boff,
                                          int wm, int wn, int l15, int hi,
                                          v8f acc[4]) {
  const __bf16* As = (const __bf16*)(smem + aoff);
  const __bf16* Bs = (const __bf16*)(smem + boff);
  // A operand: lane<16 K{0..15}, lane>=16 K{16..31}
  v16bf a;
  {
    const __bf16* ar = As + (wm * 16 + l15) * LDSROW_E + hi * 16;
    #pragma unroll
    for (int i = 0; i < 8; ++i) { a[i] = ar[i]; a[8 + i] = ar[8 + i]; }
  }
  #pragma unroll
  for (int j = 0; j < 4; ++j) {
    const __bf16* br = Bs + (wn * 64 + j * 16 + l15) * LDSROW_E + hi * 16;
    v16bf b;
    #pragma unroll
    for (int i = 0; i < 16; ++i) b[i] = br[i];
    acc[j] = wmma_bf16(a, b, acc[j]);
  }
}

__device__ __forceinline__ void gemm_body(const __bf16* __restrict__ A,
                                          const __bf16* __restrict__ Wt,
                                          int m0, int n0, int Kdim, int Arows_total,
                                          int Wrows_total, char* smem,
                                          int wave, int wm, int wn, int l15, int hi,
                                          v8f acc[4]) {
  const int nsteps = Kdim / TILE_KE;     // 24 for K=768 (even)
  const unsigned arows = (unsigned)(Arows_total - m0);
  const unsigned wrows = (unsigned)(Wrows_total - n0);
  const __bf16* Abase = A + (size_t)m0 * Kdim;
  const __bf16* Bbase = Wt + (size_t)n0 * Kdim;

  if (wave == 0) {
    tdm_load_2d(A_OFF0, Abase, arows, 32, (unsigned)Kdim, (unsigned)Kdim);
    tdm_load_2d(B_OFF0, Bbase, wrows, 256, (unsigned)Kdim, (unsigned)Kdim);
    tdm_load_2d(A_OFF1, Abase + TILE_KE, arows, 32, (unsigned)(Kdim - TILE_KE), (unsigned)Kdim);
    tdm_load_2d(B_OFF1, Bbase + TILE_KE, wrows, 256, (unsigned)(Kdim - TILE_KE), (unsigned)Kdim);
  }

  for (int kt = 0; kt < nsteps; kt += 2) {
    // ---- ping: consume buf0 ----
    __builtin_amdgcn_s_wait_tensorcnt(2);   // issuer: oldest A/B pair done; others: no-op
    __syncthreads();
    gemm_step(smem, A_OFF0, B_OFF0, wm, wn, l15, hi, acc);
    __syncthreads();
    if (wave == 0 && kt + 2 < nsteps) {
      const int kk = (kt + 2) * TILE_KE;
      tdm_load_2d(A_OFF0, Abase + kk, arows, 32, (unsigned)(Kdim - kk), (unsigned)Kdim);
      tdm_load_2d(B_OFF0, Bbase + kk, wrows, 256, (unsigned)(Kdim - kk), (unsigned)Kdim);
    }
    // ---- pong: consume buf1 ----
    __builtin_amdgcn_s_wait_tensorcnt(2);
    __syncthreads();
    gemm_step(smem, A_OFF1, B_OFF1, wm, wn, l15, hi, acc);
    __syncthreads();
    if (wave == 0 && kt + 3 < nsteps) {
      const int kk = (kt + 3) * TILE_KE;
      tdm_load_2d(A_OFF1, Abase + kk, arows, 32, (unsigned)(Kdim - kk), (unsigned)Kdim);
      tdm_load_2d(B_OFF1, Bbase + kk, wrows, 256, (unsigned)(Kdim - kk), (unsigned)Kdim);
    }
  }
}

// ---------------------------------------------------------------------------
// Kernel 1: QKV projection GEMM (M=25104, N=2304, K=768), epilogue routes to
// qh (scaled), kh, and transposed vT, all bf16. Grid: 785 M-tiles x 9 N-tiles.
// ---------------------------------------------------------------------------
__global__ __launch_bounds__(256)
void qkv_gemm_kernel(const __bf16* __restrict__ xb, const __bf16* __restrict__ Wb,
                     const float* __restrict__ bias,
                     __bf16* __restrict__ qh, __bf16* __restrict__ kh,
                     __bf16* __restrict__ vT) {
  extern __shared__ char smem[];
  const int lane = threadIdx.x & 31, wave = threadIdx.x >> 5;
  const int wm = wave >> 2, wn = wave & 3;
  const int mt = blockIdx.x % 785, nt = blockIdx.x / 785;
  const int m0 = mt * 32, n0 = nt * 256;
  const int l15 = lane & 15, hi = lane >> 4;

  v8f acc[4] = {};
  gemm_body(xb, Wb, m0, n0, D_MODEL, MROWS, 3 * D_MODEL, smem,
            wave, wm, wn, l15, hi, acc);

  #pragma unroll
  for (int j = 0; j < 4; ++j) {
    const int n   = n0 + wn * 64 + j * 16 + l15;
    const int sec = n / D_MODEL;           // 0=q 1=k 2=v
    const int nn  = n % D_MODEL;
    const int h = nn >> 6, d = nn & 63;
    const float bv = bias[n];
    #pragma unroll
    for (int r = 0; r < 8; ++r) {
      const int m = m0 + wm * 16 + r + hi * 8;
      if (m < MROWS) {
        const int bb = m / G_TOK, g = m % G_TOK;
        const size_t bh = (size_t)(bb * NH + h);
        const float v = acc[j][r] + bv;
        if (sec == 0)      qh[(bh * G_TOK + g) * HD + d] = (__bf16)(v * QSCALE);
        else if (sec == 1) kh[(bh * G_TOK + g) * HD + d] = (__bf16)v;
        else               vT[(bh * HD + d) * G_TOK + g] = (__bf16)v;
      }
    }
  }
}

// ---------------------------------------------------------------------------
// Kernel 2: windowed patch attention. One wave per (bh, frame, 16-query tile).
// Keys: t=0 -> CLS (g=0); t in [1,196] -> g = f*196 + t. 197 valid keys.
// ---------------------------------------------------------------------------
__global__ __launch_bounds__(32)
void attn_patch_kernel(const __bf16* __restrict__ qh, const __bf16* __restrict__ kh,
                       const __bf16* __restrict__ vT, __bf16* __restrict__ attn_out) {
  __shared__ float  s_s[16][208];
  __shared__ __bf16 s_p[16][224];

  const int lane = threadIdx.x;
  const int l15 = lane & 15, hi = lane >> 4;
  const int qt = blockIdx.x % 13;
  const int f  = (blockIdx.x / 13) % NF;
  const int bh = blockIdx.x / (13 * NF);
  const int b  = bh / NH, h = bh % NH;

  const int q_m = qt * 16 + l15;
  const int qg  = (q_m < NP) ? (1 + f * NP + q_m) : (1 + f * NP);
  const __bf16* qrow = qh + ((size_t)bh * G_TOK + qg) * HD;
  v16bf aq0, aq1;
  {
    const int kb = hi * 16;
    #pragma unroll
    for (int i = 0; i < 8; ++i) {
      aq0[i] = qrow[kb + i];       aq0[8 + i] = qrow[8 + kb + i];
      aq1[i] = qrow[32 + kb + i];  aq1[8 + i] = qrow[40 + kb + i];
    }
  }

  const int kb2 = hi * 16;
  for (int j = 0; j < 13; ++j) {
    const int t  = j * 16 + l15;
    const int tc = (t > NP) ? NP : t;
    const int kg = (t == 0) ? 0 : (f * NP + tc);
    const __bf16* krow = kh + ((size_t)bh * G_TOK + kg) * HD;
    v16bf b0, b1;
    #pragma unroll
    for (int i = 0; i < 16; ++i) { b0[i] = krow[kb2 + i]; b1[i] = krow[32 + kb2 + i]; }
    v8f c = {};
    c = wmma_bf16(aq0, b0, c);
    c = wmma_bf16(aq1, b1, c);
    #pragma unroll
    for (int r = 0; r < 8; ++r) s_s[r + hi * 8][j * 16 + l15] = c[r];
  }
  __syncthreads();

  const int row = lane >> 1, half = lane & 1;
  float mx = -1e30f;
  for (int ci = 0; ci < 104; ++ci) {
    const int col = half * 104 + ci;
    const float v = (col < 197) ? s_s[row][col] : -1e30f;
    mx = fmaxf(mx, v);
  }
  mx = fmaxf(mx, __shfl_xor(mx, 1, 32));
  float sum = 0.f;
  for (int ci = 0; ci < 104; ++ci) {
    const int col = half * 104 + ci;
    const float e = (col < 197) ? __expf(s_s[row][col] - mx) : 0.f;
    s_s[row][col] = e;
    sum += e;
  }
  sum += __shfl_xor(sum, 1, 32);
  const float inv = 1.f / sum;
  for (int ci = 0; ci < 104; ++ci) {
    const int col = half * 104 + ci;
    s_p[row][col] = (__bf16)(s_s[row][col] * inv);
  }
  if (half == 0) {
    #pragma unroll
    for (int i = 0; i < 16; ++i) s_p[row][208 + i] = (__bf16)0.f;
  }
  __syncthreads();

  v8f o[4] = {};
  for (int ks = 0; ks < 7; ++ks) {
    v16bf ap;
    {
      const int kb = hi * 16;
      #pragma unroll
      for (int i = 0; i < 8; ++i) {
        ap[i]     = s_p[l15][ks * 32 + kb + i];
        ap[8 + i] = s_p[l15][ks * 32 + 8 + kb + i];
      }
    }
    #pragma unroll
    for (int jd = 0; jd < 4; ++jd) {
      const int d = jd * 16 + l15;
      const __bf16* vrow = vT + ((size_t)bh * HD + d) * G_TOK;
      v16bf bv;
      #pragma unroll
      for (int i = 0; i < 16; ++i) {
        const int t  = ks * 32 + kb2 + i;
        const int tc = (t > NP) ? NP : t;
        const int gi = (t == 0) ? 0 : (f * NP + tc);
        bv[i] = vrow[gi];
      }
      o[jd] = wmma_bf16(ap, bv, o[jd]);
    }
  }

  #pragma unroll
  for (int jd = 0; jd < 4; ++jd) {
    const int d = jd * 16 + l15;
    #pragma unroll
    for (int r = 0; r < 8; ++r) {
      const int qm = qt * 16 + r + hi * 8;
      if (qm < NP) {
        const int g = 1 + f * NP + qm;
        attn_out[((size_t)b * G_TOK + g) * D_MODEL + h * HD + d] = (__bf16)o[jd][r];
      }
    }
  }
}

// ---------------------------------------------------------------------------
// Kernel 3: CLS-token attention over the full grid (192 rows, tiny) — VALU.
// ---------------------------------------------------------------------------
__global__ __launch_bounds__(256)
void attn_cls_kernel(const __bf16* __restrict__ qh, const __bf16* __restrict__ kh,
                     const __bf16* __restrict__ vT, __bf16* __restrict__ attn_out) {
  __shared__ float sc[G_TOK];
  __shared__ float qv[HD];
  __shared__ float red[256];
  const int bh = blockIdx.x, tid = threadIdx.x;
  const int b = bh / NH, h = bh % NH;

  if (tid < HD) qv[tid] = (float)qh[(size_t)bh * G_TOK * HD + tid];
  __syncthreads();

  for (int g = tid; g < G_TOK; g += 256) {
    const __bf16* kr = kh + ((size_t)bh * G_TOK + g) * HD;
    float s = 0.f;
    for (int d = 0; d < HD; ++d) s += qv[d] * (float)kr[d];
    sc[g] = s;
  }
  __syncthreads();

  float mx = -1e30f;
  for (int g = tid; g < G_TOK; g += 256) mx = fmaxf(mx, sc[g]);
  red[tid] = mx; __syncthreads();
  for (int s = 128; s > 0; s >>= 1) { if (tid < s) red[tid] = fmaxf(red[tid], red[tid + s]); __syncthreads(); }
  const float mxv = red[0];
  __syncthreads();

  float sum = 0.f;
  for (int g = tid; g < G_TOK; g += 256) { const float e = __expf(sc[g] - mxv); sc[g] = e; sum += e; }
  red[tid] = sum; __syncthreads();
  for (int s = 128; s > 0; s >>= 1) { if (tid < s) red[tid] += red[tid + s]; __syncthreads(); }
  const float inv = 1.f / red[0];
  __syncthreads();

  const int d = tid & 63, part = tid >> 6;
  const __bf16* vrow = vT + ((size_t)bh * HD + d) * G_TOK;
  float acc = 0.f;
  for (int g = part; g < G_TOK; g += 4) acc += sc[g] * (float)vrow[g];
  red[tid] = acc; __syncthreads();
  if (tid < HD) {
    const float tot = red[tid] + red[tid + 64] + red[tid + 128] + red[tid + 192];
    attn_out[(size_t)b * G_TOK * D_MODEL + h * HD + tid] = (__bf16)(tot * inv);
  }
}

// ---------------------------------------------------------------------------
// Kernel 4: output projection GEMM (M=25104, N=768, K=768), f32 out + bias.
// Grid: 785 M-tiles x 3 N-tiles.
// ---------------------------------------------------------------------------
__global__ __launch_bounds__(256)
void out_gemm_kernel(const __bf16* __restrict__ ao, const __bf16* __restrict__ Wb,
                     const float* __restrict__ bias, float* __restrict__ out) {
  extern __shared__ char smem[];
  const int lane = threadIdx.x & 31, wave = threadIdx.x >> 5;
  const int wm = wave >> 2, wn = wave & 3;
  const int mt = blockIdx.x % 785, nt = blockIdx.x / 785;
  const int m0 = mt * 32, n0 = nt * 256;
  const int l15 = lane & 15, hi = lane >> 4;

  v8f acc[4] = {};
  gemm_body(ao, Wb, m0, n0, D_MODEL, MROWS, D_MODEL, smem,
            wave, wm, wn, l15, hi, acc);

  #pragma unroll
  for (int j = 0; j < 4; ++j) {
    const int n = n0 + wn * 64 + j * 16 + l15;
    const float bv = bias[n];
    #pragma unroll
    for (int r = 0; r < 8; ++r) {
      const int m = m0 + wm * 16 + r + hi * 8;
      if (m < MROWS) out[(size_t)m * D_MODEL + n] = acc[j][r] + bv;
    }
  }
}

// ---------------------------------------------------------------------------
extern "C" void kernel_launch(void* const* d_in, const int* in_sizes, int n_in,
                              void* d_out, int out_size, void* d_ws, size_t ws_size,
                              hipStream_t stream) {
  const float* x    = (const float*)d_in[0];
  const float* win  = (const float*)d_in[1];
  const float* bin  = (const float*)d_in[2];
  const float* wout = (const float*)d_in[3];
  const float* bout = (const float*)d_in[4];
  (void)in_sizes; (void)n_in; (void)out_size; (void)ws_size;

  char* ws = (char*)d_ws;
  size_t off = 0;
  const size_t hsz = (size_t)BHEADS * G_TOK * HD * sizeof(__bf16);     // 38.56 MB
  __bf16* qh  = (__bf16*)(ws + off); off += hsz;
  __bf16* kh  = (__bf16*)(ws + off); off += hsz;
  __bf16* vT  = (__bf16*)(ws + off); off += hsz;
  __bf16* ao  = (__bf16*)(ws + off); off += (size_t)MROWS * D_MODEL * 2;
  __bf16* xb  = (__bf16*)(ws + off); off += (size_t)MROWS * D_MODEL * 2;
  __bf16* wqb = (__bf16*)(ws + off); off += (size_t)3 * D_MODEL * D_MODEL * 2;
  __bf16* wob = (__bf16*)(ws + off); off += (size_t)D_MODEL * D_MODEL * 2;

  const int nx4 = MROWS * D_MODEL / 4;
  const int nq4 = 3 * D_MODEL * D_MODEL / 4;
  const int no4 = D_MODEL * D_MODEL / 4;
  cvt_bf16_kernel<<<dim3((nx4 + 255) / 256), dim3(256), 0, stream>>>(x, xb, nx4);
  cvt_bf16_kernel<<<dim3((nq4 + 255) / 256), dim3(256), 0, stream>>>(win, wqb, nq4);
  cvt_bf16_kernel<<<dim3((no4 + 255) / 256), dim3(256), 0, stream>>>(wout, wob, no4);

  qkv_gemm_kernel<<<dim3(785 * 9), dim3(256), GEMM_LDS_BYTES, stream>>>(
      xb, wqb, bin, qh, kh, vT);
  attn_patch_kernel<<<dim3(BHEADS * NF * 13), dim3(32), 0, stream>>>(qh, kh, vT, ao);
  attn_cls_kernel<<<dim3(BHEADS), dim3(256), 0, stream>>>(qh, kh, vT, ao);
  out_gemm_kernel<<<dim3(785 * 3), dim3(256), GEMM_LDS_BYTES, stream>>>(
      ao, wob, bout, (float*)d_out);
}
